// STgramAttentionCapsuleNet_76278619177321
// MI455X (gfx1250) — matrix operand order repaired and checked by reading, not compiled, not run on hardware
//
#include <hip/hip_runtime.h>
#include <math.h>

typedef __attribute__((ext_vector_type(16))) _Float16 v16h;
typedef __attribute__((ext_vector_type(8)))  _Float16 h8;
typedef __attribute__((ext_vector_type(8)))  float    v8f;

// ---------------------------------------------------------------------------
// WMMA helpers (CDNA5 v_wmma_f32_16x16x32_f16, wave32)
// ---------------------------------------------------------------------------
__device__ __forceinline__ v8f wmma32(v16h a, v16h b, v8f c) {
  return __builtin_amdgcn_wmma_f32_16x16x32_f16(
      /*neg_a=*/false, a, /*neg_b=*/false, b,
      /*c_mod=*/(short)0, c, /*reuse_a=*/false, /*reuse_b=*/false);
}

__device__ __forceinline__ v16h cat16(h8 lo, h8 hi) {
  return __builtin_shufflevector(lo, hi, 0,1,2,3,4,5,6,7,8,9,10,11,12,13,14,15);
}
// A fragment from a contiguous row: halves 0..7 <- p[0..7], 8..15 <- p[16..23]
__device__ __forceinline__ v16h ldA(const _Float16* p) {
  h8 lo = *(const h8*)(p);
  h8 hi = *(const h8*)(p + 16);
  return cat16(lo, hi);
}
// B fragment at weight row n, offset k0 + kb: halves 0..7 <- p[0..7], 8..15 <- p[16..23]
__device__ __forceinline__ v16h ld16(const _Float16* p) {
  h8 lo = *(const h8*)(p);
  h8 hi = *(const h8*)(p + 16);
  return cat16(lo, hi);
}

// ---------------------------------------------------------------------------
// Fast transcendentals: CDNA5 has hardware TANH/EXP/RCP (TRANS32 ops)
// ---------------------------------------------------------------------------
__device__ __forceinline__ float fast_tanh(float x) {
#if __has_builtin(__builtin_amdgcn_tanhf)
  return __builtin_amdgcn_tanhf(x);        // v_tanh_f32
#else
  return tanhf(x);
#endif
}
__device__ __forceinline__ float fast_sigmoid(float x) {
  // 1/(1+exp(-x)); exp(-x) = exp2(-x*log2(e)) -> v_exp_f32 + v_rcp_f32
  float e = __builtin_amdgcn_exp2f(-x * 1.44269504088896340736f);
  return __builtin_amdgcn_rcpf(1.f + e);
}

// ---------------------------------------------------------------------------
// Prep kernels
// ---------------------------------------------------------------------------
__global__ void k_cvt(const float* __restrict__ s, _Float16* __restrict__ d, int n) {
  int i = blockIdx.x * blockDim.x + threadIdx.x;
  if (i < n) d[i] = (_Float16)s[i];
}

// wave (64, 262144) f32 -> f16 with 512 zero-pad each side: wavep[64][263168]
__global__ void k_padwave(const float* __restrict__ wave, _Float16* __restrict__ wp) {
  int i = blockIdx.x * blockDim.x + threadIdx.x;
  if (i < 64 * 263168) {
    int b = i / 263168, t = i - b * 263168;
    wp[i] = (t >= 512 && t < 262656) ? (_Float16)wave[(long)b * 262144 + (t - 512)]
                                     : (_Float16)0.f;
  }
}

// conv2 weights: c2_w[oc][ic][dk] -> w2r[oc][k], k = dk*64 + ic
__global__ void k_w2r(const float* __restrict__ s, _Float16* __restrict__ d) {
  int i = blockIdx.x * blockDim.x + threadIdx.x;
  if (i < 64 * 192) {
    int n = i / 192, k = i - n * 192;
    int dk = k >> 6, ic = k & 63;
    d[i] = (_Float16)s[n * 192 + ic * 3 + dk];
  }
}

// fc weights: fc_w[oc][ic][dk] -> fcwr[oc][k], k = dk*192 + ic
__global__ void k_fcwr(const float* __restrict__ s, _Float16* __restrict__ d) {
  int i = blockIdx.x * blockDim.x + threadIdx.x;
  if (i < 64 * 960) {
    int n = i / 960, k = i - n * 960;
    int dk = k / 192, ic = k - dk * 192;
    d[i] = (_Float16)s[n * 960 + ic * 5 + dk];
  }
}

// Stacked CfC second-stage weights: rows 0..31 = W1, 32..63 = W2, 64..95 = Wa+Wt
__global__ void k_stack(const float* __restrict__ W1, const float* __restrict__ W2,
                        const float* __restrict__ Wa, const float* __restrict__ Wt,
                        const float* __restrict__ ba, const float* __restrict__ bt,
                        _Float16* __restrict__ Ws, float* __restrict__ bat) {
  int i = blockIdx.x * blockDim.x + threadIdx.x;
  if (i < 96 * 128) {
    int j = i >> 7, k = i & 127;
    float v;
    if (j < 32)      v = W1[j * 128 + k];
    else if (j < 64) v = W2[(j - 32) * 128 + k];
    else             v = Wa[(j - 64) * 128 + k] + Wt[(j - 64) * 128 + k];
    Ws[i] = (_Float16)v;
  }
  if (i < 32) bat[i] = ba[i] + bt[i];
}

// W_caps (4,32,16) -> WcT [n=64][k=32], n = cap*16 + d
__global__ void k_capsT(const float* __restrict__ Wc, _Float16* __restrict__ WcT) {
  int i = blockIdx.x * blockDim.x + threadIdx.x;
  if (i < 64 * 32) {
    int n = i >> 5, k = i & 31;
    int cap = n >> 4, d = n & 15;
    WcT[i] = (_Float16)Wc[cap * 512 + k * 16 + d];
  }
}

// zero pad rows of x1t (rows 0 and 514 per batch)
__global__ void k_pad1(_Float16* __restrict__ x1t) {
  int i = blockIdx.x * blockDim.x + threadIdx.x;
  if (i < 64 * 128) {
    int b = i >> 7, j = i & 127;
    int row = (j < 64) ? 0 : 514;
    int col = j & 63;
    x1t[((long)b * 515 + row) * 64 + col] = (_Float16)0.f;
  }
}

// zero pad rows of combt (rows 0,1,514,515 per batch)
__global__ void k_pad2(_Float16* __restrict__ combt) {
  int i = blockIdx.x * blockDim.x + threadIdx.x;
  if (i < 64 * 768) {
    int b = i / 768, j = i - b * 768;
    int r4 = j / 192, col = j - r4 * 192;
    int row = (r4 < 2) ? r4 : 512 + r4;
    combt[((long)b * 516 + row) * 192 + col] = (_Float16)0.f;
  }
}

// ---------------------------------------------------------------------------
// conv1: GEMM rows r=(b,t') 32832, K=1024, N=64; A from padded f16 wave.
// Output transposed+padded: x1t[b][1+t][oc] (515 rows).
// ---------------------------------------------------------------------------
__global__ __launch_bounds__(32) void k_conv1(
    const _Float16* __restrict__ wavep, const _Float16* __restrict__ w1h,
    const float* __restrict__ c1b, const float* __restrict__ bg,
    const float* __restrict__ bbeta, _Float16* __restrict__ x1t) {
  const int lane = threadIdx.x;
  const int m = lane & 15, hf = lane >> 4;
  const int akb = hf ? 8 : 0, kb = hf ? 8 : 0;
  const int row = blockIdx.x * 16 + m;
  const int b = row / 513, tp = row - b * 513;
  const _Float16* ap = wavep + (long)b * 263168 + (long)tp * 512;
  v8f acc[4] = {};
  for (int ks = 0; ks < 32; ++ks) {
    const int k0 = ks * 32;
    __builtin_prefetch(ap + k0 + 128);  // global_prefetch_b8
    v16h a = ldA(ap + k0 + akb);
    const int koff = k0 + kb;
#pragma unroll
    for (int nt = 0; nt < 4; ++nt)
      acc[nt] = wmma32(a, ld16(w1h + (nt * 16 + m) * 1024 + koff), acc[nt]);
  }
  const float rsq = rsqrtf(1.00001f);
#pragma unroll
  for (int nt = 0; nt < 4; ++nt) {
    int oc = nt * 16 + m;
    float gg = bg[oc] * rsq, be = bbeta[oc], cb = c1b[oc];
#pragma unroll
    for (int r = 0; r < 8; ++r) {
      int mr = r + hf * 8;
      int grow = blockIdx.x * 16 + mr;
      int gb = grow / 513, gt = grow - gb * 513;
      float v = acc[nt][r] + cb;
      v = gg * v + be;
      v = v > 0.f ? v : 0.2f * v;
      x1t[((long)gb * 515 + 1 + gt) * 64 + oc] = (_Float16)v;
    }
  }
}

// ---------------------------------------------------------------------------
// conv2: K=192 (k = dk*64+ic), A chunks contiguous from x1t. Out x2[b][64][513].
// ---------------------------------------------------------------------------
__global__ __launch_bounds__(32) void k_conv2(
    const _Float16* __restrict__ x1t, const _Float16* __restrict__ w2r,
    const float* __restrict__ c2b, const float* __restrict__ bg,
    const float* __restrict__ bbeta, _Float16* __restrict__ x2h) {
  const int lane = threadIdx.x;
  const int m = lane & 15, hf = lane >> 4;
  const int akb = hf ? 8 : 0, kb = hf ? 8 : 0;
  const int row = blockIdx.x * 16 + m;
  const int b = row / 513, tp = row - b * 513;
  const _Float16* xb = x1t + ((long)b * 515 + tp) * 64;  // row t+dk -> +dk*64
  v8f acc[4] = {};
#pragma unroll
  for (int ks = 0; ks < 6; ++ks) {
    const int k0 = ks * 32;
    const int ka = k0 + akb, kc = ka + 16;
    h8 lo = *(const h8*)(xb + (ka >> 6) * 64 + (ka & 63));
    h8 hi = *(const h8*)(xb + (kc >> 6) * 64 + (kc & 63));
    v16h a = cat16(lo, hi);
    const int koff = k0 + kb;
#pragma unroll
    for (int nt = 0; nt < 4; ++nt)
      acc[nt] = wmma32(a, ld16(w2r + (nt * 16 + m) * 192 + koff), acc[nt]);
  }
  const float rsq = rsqrtf(1.00001f);
#pragma unroll
  for (int nt = 0; nt < 4; ++nt) {
    int oc = nt * 16 + m;
    float gg = bg[oc] * rsq, be = bbeta[oc], cb = c2b[oc];
#pragma unroll
    for (int r = 0; r < 8; ++r) {
      int mr = r + hf * 8;
      int grow = blockIdx.x * 16 + mr;
      int gb = grow / 513, gt = grow - gb * 513;
      float v = acc[nt][r] + cb;
      v = gg * v + be;
      v = v > 0.f ? v : 0.2f * v;
      x2h[((long)gb * 64 + oc) * 513 + gt] = (_Float16)v;
    }
  }
}

// ---------------------------------------------------------------------------
// combined: ic<128 -> sgram[b][t][ic]; else interp(x2, 513->512).
// Writes combt[b][2+t][ic] (t-major, padded) + channel means.
// ---------------------------------------------------------------------------
__global__ __launch_bounds__(32) void k_comb(const float* __restrict__ sgram,
                                             const _Float16* __restrict__ x2h,
                                             _Float16* __restrict__ combt,
                                             float* __restrict__ mmean) {
  const int bc = blockIdx.x;             // b*192 + ic
  const int b = bc / 192, ic = bc - b * 192;
  const int lane = threadIdx.x;
  float sum = 0.f;
  for (int i = 0; i < 16; ++i) {
    int t = lane + i * 32;
    float val;
    if (ic < 128) {
      val = sgram[((long)b * 512 + t) * 128 + ic];
    } else {
      int c = ic - 128;
      float pos = (t + 0.5f) * (513.f / 512.f) - 0.5f;
      pos = fminf(fmaxf(pos, 0.f), 512.f);
      int lo = (int)floorf(pos);
      int hi = lo + 1 < 513 ? lo + 1 : 512;
      float w = pos - (float)lo;
      const _Float16* xr = x2h + ((long)b * 64 + c) * 513;
      val = (float)xr[lo] * (1.f - w) + (float)xr[hi] * w;
    }
    combt[((long)b * 516 + 2 + t) * 192 + ic] = (_Float16)val;
    sum += val;
  }
  for (int off = 16; off > 0; off >>= 1) sum += __shfl_down(sum, off, 32);
  if (lane == 0) mmean[bc] = sum * (1.f / 512.f);
}

// ---------------------------------------------------------------------------
// Squeeze-excite gate
// ---------------------------------------------------------------------------
__global__ __launch_bounds__(64) void k_gate(const float* __restrict__ m,
                                             const float* __restrict__ g1w,
                                             const float* __restrict__ g1b,
                                             const float* __restrict__ g2w,
                                             const float* __restrict__ g2b,
                                             float* __restrict__ g) {
  const int b = blockIdx.x, tid = threadIdx.x;
  __shared__ float ms[192];
  __shared__ float h48[48];
  for (int i = tid; i < 192; i += 64) ms[i] = m[b * 192 + i];
  __syncthreads();
  if (tid < 48) {
    float s = g1b[tid];
    for (int ic = 0; ic < 192; ++ic) s += ms[ic] * g1w[tid * 192 + ic];
    h48[tid] = fmaxf(s, 0.f);
  }
  __syncthreads();
  for (int o = tid; o < 192; o += 64) {
    float s = g2b[o];
    for (int j = 0; j < 48; ++j) s += h48[j] * g2w[o * 48 + j];
    g[b * 192 + o] = fast_sigmoid(s);
  }
}

// fused = combined * g, in place on combt interior (regenerated every call)
__global__ __launch_bounds__(192) void k_fuse(_Float16* __restrict__ combt,
                                              const float* __restrict__ g) {
  const int bt = blockIdx.x;             // b*512 + t
  const int b = bt >> 9, t = bt & 511;
  const int ic = threadIdx.x;
  long idx = ((long)b * 516 + 2 + t) * 192 + ic;
  combt[idx] = (_Float16)((float)combt[idx] * g[b * 192 + ic]);
}

// ---------------------------------------------------------------------------
// fc conv: K=960 (k = dk*192+ic), A chunks contiguous from padded combt.
// relu -> feat[(b*512+t)][64]
// ---------------------------------------------------------------------------
__global__ __launch_bounds__(32) void k_fcconv(
    const _Float16* __restrict__ combt, const _Float16* __restrict__ fcwr,
    const float* __restrict__ fcb, _Float16* __restrict__ feath) {
  const int lane = threadIdx.x;
  const int m = lane & 15, hf = lane >> 4;
  const int akb = hf ? 8 : 0, kb = hf ? 8 : 0;
  const int row = blockIdx.x * 16 + m;
  const int b = row >> 9, t = row & 511;
  const _Float16* xb = combt + ((long)b * 516 + t) * 192;  // row t+dk -> +dk*192
  v8f acc[4] = {};
  for (int ks = 0; ks < 30; ++ks) {
    const int k0 = ks * 32;
    const int ka = k0 + akb, kc = ka + 16;
    const int dka = ka / 192, dkc = kc / 192;
    h8 lo = *(const h8*)(xb + dka * 192 + (ka - dka * 192));
    h8 hi = *(const h8*)(xb + dkc * 192 + (kc - dkc * 192));
    v16h a = cat16(lo, hi);
    const int koff = k0 + kb;
#pragma unroll
    for (int nt = 0; nt < 4; ++nt)
      acc[nt] = wmma32(a, ld16(fcwr + (nt * 16 + m) * 960 + koff), acc[nt]);
  }
#pragma unroll
  for (int nt = 0; nt < 4; ++nt) {
    int oc = nt * 16 + m;
    float bias = fcb[oc];
#pragma unroll
    for (int r = 0; r < 8; ++r) {
      int mr = r + hf * 8;
      int grow = blockIdx.x * 16 + mr;
      float v = fmaxf(acc[nt][r] + bias, 0.f);
      feath[(long)grow * 64 + oc] = (_Float16)v;
    }
  }
}

// ---------------------------------------------------------------------------
// CfC recurrence: 1 wave / 16 batch rows, T=512 steps, weights+acts in LDS.
// GEMM1: Xcat(16x96)@Wb^T -> Z(16x128) lecun_tanh
// GEMM2: Z(16x128)@[W1;W2;Wa+Wt]^T -> tanh/tanh/sigmoid combine -> h
// ---------------------------------------------------------------------------
__global__ __launch_bounds__(32) void k_cfc(
    const _Float16* __restrict__ xin,  // [B*T][64]
    const _Float16* __restrict__ Wbh,  // [128][96]
    const float* __restrict__ bb,      // [128]
    const _Float16* __restrict__ Wsh,  // [96][128] stacked
    const float* __restrict__ b1, const float* __restrict__ b2,
    const float* __restrict__ bat,
    _Float16* __restrict__ outh /* [B*T][32] */) {
  __shared__ __align__(16) _Float16 Wb_s[128 * 96];
  __shared__ __align__(16) _Float16 Ws_s[96 * 128];
  __shared__ __align__(16) _Float16 X_s[16 * 96];   // [xt(64) | h(32)]
  __shared__ __align__(16) _Float16 Z_s[16 * 128];
  const int lane = threadIdx.x;
  for (int i = lane; i < 1536; i += 32) {
    ((h8*)Wb_s)[i] = ((const h8*)Wbh)[i];
    ((h8*)Ws_s)[i] = ((const h8*)Wsh)[i];
  }
  for (int i = lane; i < 16 * 32; i += 32)
    X_s[(i >> 5) * 96 + 64 + (i & 31)] = (_Float16)0.f;
  __syncthreads();

  const int m = lane & 15, hf = lane >> 4;
  const int akb = hf ? 8 : 0, kb = hf ? 8 : 0;
  float bbn[8];
#pragma unroll
  for (int nt = 0; nt < 8; ++nt) bbn[nt] = bb[nt * 16 + m];
  float c1b_[2], c2b_[2], cab_[2];
#pragma unroll
  for (int p = 0; p < 2; ++p) {
    int j = p * 16 + m;
    c1b_[p] = b1[j]; c2b_[p] = b2[j]; cab_[p] = bat[j];
  }
  const int bbase = blockIdx.x * 16;

  for (int t = 0; t < 512; ++t) {
    for (int i = lane; i < 128; i += 32) {
      int rr = i >> 3, c8 = i & 7;
      ((h8*)(X_s + rr * 96))[c8] =
          ((const h8*)(xin + ((long)(bbase + rr) * 512 + t) * 64))[c8];
    }
    __syncthreads();

    v8f acc[8] = {};
#pragma unroll
    for (int ks = 0; ks < 3; ++ks) {
      v16h a = ldA(X_s + m * 96 + ks * 32 + akb);
      const int koff = ks * 32 + kb;
#pragma unroll
      for (int nt = 0; nt < 8; ++nt)
        acc[nt] = wmma32(a, ld16(Wb_s + (nt * 16 + m) * 96 + koff), acc[nt]);
    }
#pragma unroll
    for (int nt = 0; nt < 8; ++nt) {
      int n = nt * 16 + m;
#pragma unroll
      for (int r = 0; r < 8; ++r) {
        int mr = r + hf * 8;
        float z = acc[nt][r] + bbn[nt];
        z = 1.7159f * fast_tanh(0.666f * z);
        Z_s[mr * 128 + n] = (_Float16)z;
      }
    }
    __syncthreads();

    v8f acc2[6] = {};
#pragma unroll
    for (int ks = 0; ks < 4; ++ks) {
      v16h a = ldA(Z_s + m * 128 + ks * 32 + akb);
      const int koff = ks * 32 + kb;
#pragma unroll
      for (int nt = 0; nt < 6; ++nt)
        acc2[nt] = wmma32(a, ld16(Ws_s + (nt * 16 + m) * 128 + koff), acc2[nt]);
    }
#pragma unroll
    for (int p = 0; p < 2; ++p) {
      int j = p * 16 + m;
#pragma unroll
      for (int r = 0; r < 8; ++r) {
        int mr = r + hf * 8;
        float f1 = fast_tanh(acc2[p][r] + c1b_[p]);
        float f2 = fast_tanh(acc2[2 + p][r] + c2b_[p]);
        float ti = fast_sigmoid(acc2[4 + p][r] + cab_[p]);
        float hn = f1 * (1.f - ti) + ti * f2;
        X_s[mr * 96 + 64 + j] = (_Float16)hn;
        outh[((long)(bbase + mr) * 512 + t) * 32 + j] = (_Float16)hn;
      }
    }
    __syncthreads();
  }
}

// ---------------------------------------------------------------------------
// Capsules: enc[r][32] @ WcT -> caps[r][64], squash per 16-dim capsule
// ---------------------------------------------------------------------------
__global__ __launch_bounds__(32) void k_caps(const _Float16* __restrict__ ench,
                                             const _Float16* __restrict__ WcT,
                                             float* __restrict__ caps_out,
                                             _Float16* __restrict__ capsh) {
  const int lane = threadIdx.x;
  const int m = lane & 15, hf = lane >> 4;
  const int akb = hf ? 8 : 0, kb = hf ? 8 : 0;
  const int row = blockIdx.x * 16 + m;
  v16h a = ldA(ench + (long)row * 32 + akb);
  v8f acc[4] = {};
#pragma unroll
  for (int nt = 0; nt < 4; ++nt)
    acc[nt] = wmma32(a, ld16(WcT + (nt * 16 + m) * 32 + kb), acc[nt]);
#pragma unroll
  for (int nt = 0; nt < 4; ++nt) {
    int n = nt * 16 + m;
#pragma unroll
    for (int r = 0; r < 8; ++r) {
      float v = acc[nt][r];
      float s = v * v;
      s += __shfl_xor(s, 1, 32);
      s += __shfl_xor(s, 2, 32);
      s += __shfl_xor(s, 4, 32);
      s += __shfl_xor(s, 8, 32);
      float scale = (s / (1.f + s)) * rsqrtf(s + 1e-8f);
      float o = v * scale;
      int mr = r + hf * 8;
      long grow = blockIdx.x * 16 + mr;
      caps_out[grow * 64 + n] = o;
      capsh[grow * 64 + n] = (_Float16)o;
    }
  }
}

// ---------------------------------------------------------------------------
// recon: dec[r][32] @ o_w^T + o_b -> out[r][128]
// ---------------------------------------------------------------------------
__global__ __launch_bounds__(32) void k_recon(const _Float16* __restrict__ dech,
                                              const _Float16* __restrict__ owh,
                                              const float* __restrict__ ob,
                                              float* __restrict__ out) {
  const int lane = threadIdx.x;
  const int m = lane & 15, hf = lane >> 4;
  const int akb = hf ? 8 : 0, kb = hf ? 8 : 0;
  const int row = blockIdx.x * 16 + m;
  v16h a = ldA(dech + (long)row * 32 + akb);
  v8f acc[8] = {};
#pragma unroll
  for (int nt = 0; nt < 8; ++nt)
    acc[nt] = wmma32(a, ld16(owh + (nt * 16 + m) * 32 + kb), acc[nt]);
#pragma unroll
  for (int nt = 0; nt < 8; ++nt) {
    int n = nt * 16 + m;
    float bias = ob[n];
#pragma unroll
    for (int r = 0; r < 8; ++r) {
      int mr = r + hf * 8;
      long grow = blockIdx.x * 16 + mr;
      out[grow * 128 + n] = acc[nt][r] + bias;
    }
  }
}

// ---------------------------------------------------------------------------
// Launch
// ---------------------------------------------------------------------------
extern "C" void kernel_launch(void* const* d_in, const int* in_sizes, int n_in,
                              void* d_out, int out_size, void* d_ws, size_t ws_size,
                              hipStream_t stream) {
  (void)in_sizes; (void)n_in; (void)out_size; (void)ws_size;
  const float* sgram = (const float*)d_in[0];
  const float* wave  = (const float*)d_in[1];
  const float* c1_w  = (const float*)d_in[2];
  const float* c1_b  = (const float*)d_in[3];
  const float* bn1_g = (const float*)d_in[4];
  const float* bn1_b = (const float*)d_in[5];
  const float* c2_w  = (const float*)d_in[6];
  const float* c2_b  = (const float*)d_in[7];
  const float* bn2_g = (const float*)d_in[8];
  const float* bn2_b = (const float*)d_in[9];
  const float* g1_w  = (const float*)d_in[10];
  const float* g1_b  = (const float*)d_in[11];
  const float* g2_w  = (const float*)d_in[12];
  const float* g2_b  = (const float*)d_in[13];
  const float* fc_w  = (const float*)d_in[14];
  const float* fc_b  = (const float*)d_in[15];
  const float* e_Wb = (const float*)d_in[16];
  const float* e_bb = (const float*)d_in[17];
  const float* e_W1 = (const float*)d_in[18];
  const float* e_W2 = (const float*)d_in[19];
  const float* e_Wa = (const float*)d_in[20];
  const float* e_Wt = (const float*)d_in[21];
  const float* e_b1 = (const float*)d_in[22];
  const float* e_b2 = (const float*)d_in[23];
  const float* e_ba = (const float*)d_in[24];
  const float* e_bt = (const float*)d_in[25];
  const float* d_Wb = (const float*)d_in[26];
  const float* d_bb = (const float*)d_in[27];
  const float* d_W1 = (const float*)d_in[28];
  const float* d_W2 = (const float*)d_in[29];
  const float* d_Wa = (const float*)d_in[30];
  const float* d_Wt = (const float*)d_in[31];
  const float* d_b1 = (const float*)d_in[32];
  const float* d_b2 = (const float*)d_in[33];
  const float* d_ba = (const float*)d_in[34];
  const float* d_bt = (const float*)d_in[35];
  const float* W_caps = (const float*)d_in[36];
  const float* o_w  = (const float*)d_in[37];
  const float* o_b  = (const float*)d_in[38];

  float* out_recon = (float*)d_out;                       // 64*512*128
  float* out_caps  = out_recon + (long)64 * 512 * 128;    // 64*512*64

  char* base = (char*)d_ws;
  size_t off = 0;
  auto alloc = [&](size_t bytes) -> char* {
    off = (off + 255) & ~(size_t)255;
    char* p = base + off;
    off += bytes;
    return p;
  };
  _Float16* w1h   = (_Float16*)alloc(65536 * 2);
  _Float16* w2r   = (_Float16*)alloc(12288 * 2);
  _Float16* fcwr  = (_Float16*)alloc(61440 * 2);
  _Float16* eWbh  = (_Float16*)alloc(12288 * 2);
  _Float16* eWsh  = (_Float16*)alloc(12288 * 2);
  float*    e_bat = (float*)   alloc(32 * 4);
  _Float16* dWbh  = (_Float16*)alloc(12288 * 2);
  _Float16* dWsh  = (_Float16*)alloc(12288 * 2);
  float*    d_bat = (float*)   alloc(32 * 4);
  _Float16* WcT   = (_Float16*)alloc(2048 * 2);
  _Float16* owh   = (_Float16*)alloc(4096 * 2);
  _Float16* wavep = (_Float16*)alloc((size_t)64 * 263168 * 2);
  _Float16* x1t   = (_Float16*)alloc((size_t)64 * 515 * 64 * 2);
  _Float16* x2h   = (_Float16*)alloc((size_t)64 * 64 * 513 * 2);
  _Float16* combt = (_Float16*)alloc((size_t)64 * 516 * 192 * 2);
  float*    m_ws  = (float*)   alloc((size_t)64 * 192 * 4);
  float*    g_ws  = (float*)   alloc((size_t)64 * 192 * 4);
  _Float16* feath = (_Float16*)alloc((size_t)64 * 512 * 64 * 2);
  _Float16* ench  = (_Float16*)alloc((size_t)64 * 512 * 32 * 2);
  _Float16* capsh = (_Float16*)alloc((size_t)64 * 512 * 64 * 2);
  _Float16* dech  = (_Float16*)alloc((size_t)64 * 512 * 32 * 2);

  // --- prep: conversions, repacks, padding ---
  k_cvt<<<(65536 + 255) / 256, 256, 0, stream>>>(c1_w, w1h, 65536);
  k_w2r<<<(12288 + 255) / 256, 256, 0, stream>>>(c2_w, w2r);
  k_fcwr<<<(61440 + 255) / 256, 256, 0, stream>>>(fc_w, fcwr);
  k_cvt<<<(12288 + 255) / 256, 256, 0, stream>>>(e_Wb, eWbh, 12288);
  k_cvt<<<(12288 + 255) / 256, 256, 0, stream>>>(d_Wb, dWbh, 12288);
  k_cvt<<<(4096 + 255) / 256, 256, 0, stream>>>(o_w, owh, 4096);
  k_stack<<<48, 256, 0, stream>>>(e_W1, e_W2, e_Wa, e_Wt, e_ba, e_bt, eWsh, e_bat);
  k_stack<<<48, 256, 0, stream>>>(d_W1, d_W2, d_Wa, d_Wt, d_ba, d_bt, dWsh, d_bat);
  k_capsT<<<8, 256, 0, stream>>>(W_caps, WcT);
  k_padwave<<<(64 * 263168 + 255) / 256, 256, 0, stream>>>(wave, wavep);
  k_pad1<<<(64 * 128 + 255) / 256, 256, 0, stream>>>(x1t);
  k_pad2<<<(64 * 768 + 255) / 256, 256, 0, stream>>>(combt);

  // --- frontend ---
  k_conv1<<<2052, 32, 0, stream>>>(wavep, w1h, c1_b, bn1_g, bn1_b, x1t);
  k_conv2<<<2052, 32, 0, stream>>>(x1t, w2r, c2_b, bn2_g, bn2_b, x2h);
  k_comb<<<12288, 32, 0, stream>>>(sgram, x2h, combt, m_ws);
  k_gate<<<64, 64, 0, stream>>>(m_ws, g1_w, g1_b, g2_w, g2_b, g_ws);
  k_fuse<<<64 * 512, 192, 0, stream>>>(combt, g_ws);
  k_fcconv<<<2048, 32, 0, stream>>>(combt, fcwr, fc_b, feath);

  // --- recurrent core + heads ---
  k_cfc<<<4, 32, 0, stream>>>(feath, eWbh, e_bb, eWsh, e_b1, e_b2, e_bat, ench);
  k_caps<<<2048, 32, 0, stream>>>(ench, WcT, out_caps, capsh);
  k_cfc<<<4, 32, 0, stream>>>(capsh, dWbh, d_bb, dWsh, d_b1, d_b2, d_bat, dech);
  k_recon<<<2048, 32, 0, stream>>>(dech, owh, o_b, out_recon);
}
